// Decoder_71725953843677
// MI455X (gfx1250) — matrix operand order repaired and checked by reading, not compiled, not run on hardware
//
#include <hip/hip_runtime.h>

// ---------------------------------------------------------------------------
// ChebConv (K=5) x2 for MI455X / gfx1250.
//   - Dense GEMMs: v_wmma_f32_16x16x32_bf16, fully unrolled K (templated),
//     operands pre-converted to bf16 so inner loop is pure b128 loads + WMMA.
//   - Sparse propagation (the roofline bottleneck, ~8 GB L2-resident traffic):
//     coalesced float4 gather + f32 global atomic scatter.
//   - bias + ReLU fused into the last GEMM epilogue of each layer.
// ---------------------------------------------------------------------------

typedef __attribute__((ext_vector_type(16))) __bf16          v16bf;
typedef __attribute__((ext_vector_type(16))) unsigned short  v16u;
typedef __attribute__((ext_vector_type(8)))  unsigned short  v8u;
typedef __attribute__((ext_vector_type(8)))  float           v8f;

#define N_NODES 50000
#define N_EDGES 800000
#define FEAT_IN 64
#define FEAT_HID 256
#define FEAT_OUT 64
#define K_CHEB 5

__device__ __forceinline__ unsigned short f2bf_bits(float f) {
  unsigned u = __builtin_bit_cast(unsigned, f);
  // round-to-nearest-even truncation to bf16
  return (unsigned short)((u + 0x7FFFu + ((u >> 16) & 1u)) >> 16);
}

// --------------------------- elementwise helpers ---------------------------

__global__ void zero_f32(float* __restrict__ p, long long n) {
  long long i = (long long)blockIdx.x * blockDim.x + threadIdx.x;
  long long stride = (long long)gridDim.x * blockDim.x;
  for (; i < n; i += stride) p[i] = 0.0f;
}

// dst = 2*dst - t0   (Chebyshev recurrence combine)
__global__ void cheb_combine(float* __restrict__ dst, const float* __restrict__ t0,
                             long long n) {
  long long i = (long long)blockIdx.x * blockDim.x + threadIdx.x;
  long long stride = (long long)gridDim.x * blockDim.x;
  for (; i < n; i += stride) dst[i] = 2.0f * dst[i] - t0[i];
}

// f32 -> bf16, 4 elements per thread (n4 = count/4)
__global__ void cvt_f32_bf16(const float* __restrict__ in,
                             unsigned short* __restrict__ out, long long n4) {
  long long i = (long long)blockIdx.x * blockDim.x + threadIdx.x;
  long long stride = (long long)gridDim.x * blockDim.x;
  for (; i < n4; i += stride) {
    float4 s = ((const float4*)in)[i];
    ushort4 o;
    o.x = f2bf_bits(s.x);
    o.y = f2bf_bits(s.y);
    o.z = f2bf_bits(s.z);
    o.w = f2bf_bits(s.w);
    ((ushort4*)out)[i] = o;
  }
}

// Wt[n*fin + k] = bf16(W[k*fout + n])  (one K-slice: fin x fout, transposed)
__global__ void cvt_w_bf16_t(const float* __restrict__ W,
                             unsigned short* __restrict__ Wt, int fin, int fout) {
  int idx = blockIdx.x * blockDim.x + threadIdx.x;
  if (idx >= fin * fout) return;
  int nn = idx / fin;
  int k  = idx - nn * fin;
  Wt[idx] = f2bf_bits(W[(size_t)k * fout + nn]);
}

// --------------------------- edge normalization ----------------------------

__global__ void deg_kernel(const int* __restrict__ row, const float* __restrict__ w,
                           float* __restrict__ deg, int e) {
  int i = blockIdx.x * blockDim.x + threadIdx.x;
  if (i < e) atomicAdd(&deg[row[i]], w[i]);
}

__global__ void dis_kernel(const float* __restrict__ deg, float* __restrict__ dis, int n) {
  int i = blockIdx.x * blockDim.x + threadIdx.x;
  if (i < n) {
    float d = deg[i];
    dis[i] = (d > 0.0f) ? __frsqrt_rn(d) : 0.0f;
  }
}

__global__ void norm_kernel(const int* __restrict__ row, const int* __restrict__ col,
                            const float* __restrict__ w, const float* __restrict__ dis,
                            float* __restrict__ nrm, int e) {
  int i = blockIdx.x * blockDim.x + threadIdx.x;
  if (i < e) nrm[i] = -dis[row[i]] * w[i] * dis[col[i]];
}

// --------------------------- sparse propagation ----------------------------
// out[col[e], :] += norm[e] * in[row[e], :]
// threadIdx.x spans feature chunks of ONE edge -> contiguous float4 gather of
// the full source row; threadIdx.y spans edges.  E divides block.y exactly.
__global__ void prop_kernel(const int* __restrict__ row, const int* __restrict__ col,
                            const float* __restrict__ nrm, const float* __restrict__ in,
                            float* __restrict__ out, int e, int f) {
  int ei = blockIdx.x * blockDim.y + threadIdx.y;
  if (ei >= e) return;
  int c4 = threadIdx.x << 2;
  int r  = row[ei];
  int cl = col[ei];
  float nv = nrm[ei];
  const float4 s = *(const float4*)(in + (size_t)r * f + c4);
  float* dst = out + (size_t)cl * f + c4;
  atomicAdd(dst + 0, nv * s.x);
  atomicAdd(dst + 1, nv * s.y);
  atomicAdd(dst + 2, nv * s.z);
  atomicAdd(dst + 3, nv * s.w);
}

// ------------------------------- WMMA GEMM ---------------------------------
// out = [Cin +] A @ W (+ bias, ReLU).  A: N_NODES x FIN bf16 row-major,
// Wt: FOUT x FIN bf16 (pre-transposed).  One wave = one 16x16 output tile,
// K marched in fully-unrolled steps of 32 via v_wmma_f32_16x16x32_bf16.
template <int FIN, int FOUT>
__global__ void gemm_wmma_bf16(const unsigned short* __restrict__ A,
                               const unsigned short* __restrict__ Wt,
                               const float* __restrict__ Cin,
                               const float* __restrict__ bias,
                               float* __restrict__ out, int relu) {
  constexpr int TILES_N = FOUT / 16;
  int wave = blockIdx.x * (blockDim.x >> 5) + (threadIdx.x >> 5);
  int lane = threadIdx.x & 31;
  int tile_m = wave / TILES_N;                 // uniform per wave
  int tile_n = wave - tile_m * TILES_N;
  if (tile_m * 16 >= N_NODES) return;          // wave-uniform: EXEC stays ~0

  int half = lane >> 4;                        // 0: lanes 0-15, 1: lanes 16-31
  int l    = lane & 15;
  int colIdx = tile_n * 16 + l;
  int rowIdx = tile_m * 16 + l;

  // C/D layout: VGPR r holds row (r + 8*half), column = lane&15
  v8f c = {};
  if (Cin) {
#pragma unroll
    for (int r = 0; r < 8; ++r)
      c[r] = Cin[(size_t)(tile_m * 16 + r + half * 8) * FOUT + colIdx];
  }

  // A layout (16-bit 16x32): half 0 holds K {k0+0..7, k0+16..23},
  //                          half 1 holds K {k0+8..15, k0+24..31}
  const unsigned short* arow = A  + (size_t)rowIdx * FIN + half * 8;
  // B layout (16-bit 32x16): lane = column; same K split as A
  const unsigned short* bcol = Wt + (size_t)colIdx * FIN + half * 8;

  union V16 { v16u v; v8u h[2]; };
#pragma unroll
  for (int k0 = 0; k0 < FIN; k0 += 32) {
    V16 au, bu;
    au.h[0] = *(const v8u*)(arow + k0);        // 16B contiguous
    au.h[1] = *(const v8u*)(arow + k0 + 16);
    bu.h[0] = *(const v8u*)(bcol + k0);
    bu.h[1] = *(const v8u*)(bcol + k0 + 16);
    c = __builtin_amdgcn_wmma_f32_16x16x32_bf16(
        /*neg_a=*/false, __builtin_bit_cast(v16bf, au.v),
        /*neg_b=*/false, __builtin_bit_cast(v16bf, bu.v),
        /*c_mod=*/(short)0, c, /*reuse_a=*/false, /*reuse_b=*/false);
  }

  float bv = bias ? bias[colIdx] : 0.0f;
#pragma unroll
  for (int r = 0; r < 8; ++r) {
    float v = c[r] + bv;
    if (relu) v = fmaxf(v, 0.0f);
    out[(size_t)(tile_m * 16 + r + half * 8) * FOUT + colIdx] = v;
  }
}

// ------------------------------- host side ---------------------------------

template <int FIN, int FOUT>
static void run_cheb_layer(const float* A0, const unsigned short* Wt_bf,
                           const float* bias, float* acc,
                           float* bufA, float* bufB, float* bufC,
                           unsigned short* abf,
                           const int* row, const int* col, const float* nrm,
                           int relu, hipStream_t stream) {
  const long long feat = (long long)N_NODES * FIN;
  dim3 pblock(FIN / 4, 256 / (FIN / 4));
  dim3 pgrid((N_EDGES + pblock.y - 1) / pblock.y);
  constexpr int tiles   = (N_NODES / 16) * (FOUT / 16);
  constexpr int gblocks = (tiles + 7) / 8;     // 8 waves / 256-thread block
  const size_t wsz = (size_t)FIN * FOUT;

  // k = 0 : acc = T0 @ W[0]
  cvt_f32_bf16<<<2048, 256, 0, stream>>>(A0, abf, feat / 4);
  gemm_wmma_bf16<FIN, FOUT><<<gblocks, 256, 0, stream>>>(
      abf, Wt_bf, nullptr, nullptr, acc, 0);

  // k = 1 : T1 = L_hat @ T0 ; acc += T1 @ W[1]
  zero_f32<<<2048, 256, 0, stream>>>(bufA, feat);
  prop_kernel<<<pgrid, pblock, 0, stream>>>(row, col, nrm, A0, bufA, N_EDGES, FIN);
  cvt_f32_bf16<<<2048, 256, 0, stream>>>(bufA, abf, feat / 4);
  gemm_wmma_bf16<FIN, FOUT><<<gblocks, 256, 0, stream>>>(
      abf, Wt_bf + wsz, acc, nullptr, acc, 0);

  const float* t0 = A0;
  float* t1 = bufA;
  for (int k = 2; k < K_CHEB; ++k) {
    float* t2 = (k == 2) ? bufB : (k == 3) ? bufC : bufA;  // t2 != t0, t1
    zero_f32<<<2048, 256, 0, stream>>>(t2, feat);
    prop_kernel<<<pgrid, pblock, 0, stream>>>(row, col, nrm, t1, t2, N_EDGES, FIN);
    cheb_combine<<<2048, 256, 0, stream>>>(t2, t0, feat);  // T2 = 2*prop(T1)-T0
    cvt_f32_bf16<<<2048, 256, 0, stream>>>(t2, abf, feat / 4);
    bool last = (k == K_CHEB - 1);
    gemm_wmma_bf16<FIN, FOUT><<<gblocks, 256, 0, stream>>>(
        abf, Wt_bf + (size_t)k * wsz, acc, last ? bias : nullptr, acc,
        last ? relu : 0);
    t0 = t1;
    t1 = t2;
  }
}

extern "C" void kernel_launch(void* const* d_in, const int* in_sizes, int n_in,
                              void* d_out, int out_size, void* d_ws, size_t ws_size,
                              hipStream_t stream) {
  const float* x   = (const float*)d_in[0];            // N x 64
  const int*   ei  = (const int*)d_in[1];              // 2 x E
  const float* w   = (const float*)d_in[2];            // E
  const float* W1  = (const float*)d_in[3];            // K x 64 x 256
  const float* b1  = (const float*)d_in[4];            // 256
  const float* W2  = (const float*)d_in[5];            // K x 256 x 64
  const float* b2  = (const float*)d_in[6];            // 64
  float*       out = (float*)d_out;                    // N x 64

  const int* row = ei;
  const int* col = ei + N_EDGES;

  // ---- workspace carve (256B aligned) ----
  char* ws = (char*)d_ws;
  size_t off = 0;
  auto carveB = [&](size_t bytes) -> char* {
    char* p = ws + off;
    off += (bytes + 255) & ~(size_t)255;
    return p;
  };
  float* deg  = (float*)carveB((size_t)N_NODES * 4);
  float* dis  = (float*)carveB((size_t)N_NODES * 4);
  float* nrm  = (float*)carveB((size_t)N_EDGES * 4);
  float* bufA = (float*)carveB((size_t)N_NODES * FEAT_HID * 4);
  float* bufB = (float*)carveB((size_t)N_NODES * FEAT_HID * 4);
  float* bufC = (float*)carveB((size_t)N_NODES * FEAT_HID * 4);
  float* h    = (float*)carveB((size_t)N_NODES * FEAT_HID * 4);
  unsigned short* abf = (unsigned short*)carveB((size_t)N_NODES * FEAT_HID * 2);
  unsigned short* w1t = (unsigned short*)carveB((size_t)K_CHEB * FEAT_IN * FEAT_HID * 2);
  unsigned short* w2t = (unsigned short*)carveB((size_t)K_CHEB * FEAT_HID * FEAT_OUT * 2);
  (void)ws_size; (void)in_sizes; (void)n_in; (void)out_size;

  // ---- weights: convert + transpose to bf16 once ----
  for (int k = 0; k < K_CHEB; ++k) {
    cvt_w_bf16_t<<<(FEAT_IN * FEAT_HID + 255) / 256, 256, 0, stream>>>(
        W1 + (size_t)k * FEAT_IN * FEAT_HID,
        w1t + (size_t)k * FEAT_IN * FEAT_HID, FEAT_IN, FEAT_HID);
    cvt_w_bf16_t<<<(FEAT_HID * FEAT_OUT + 255) / 256, 256, 0, stream>>>(
        W2 + (size_t)k * FEAT_HID * FEAT_OUT,
        w2t + (size_t)k * FEAT_HID * FEAT_OUT, FEAT_HID, FEAT_OUT);
  }

  // ---- edge normalization: nrm = -dis[row] * w * dis[col] ----
  zero_f32<<<256, 256, 0, stream>>>(deg, N_NODES);
  deg_kernel<<<(N_EDGES + 255) / 256, 256, 0, stream>>>(row, w, deg, N_EDGES);
  dis_kernel<<<(N_NODES + 255) / 256, 256, 0, stream>>>(deg, dis, N_NODES);
  norm_kernel<<<(N_EDGES + 255) / 256, 256, 0, stream>>>(row, col, w, dis, nrm, N_EDGES);

  // ---- layer 1: h = relu(cheb(x, W1) + b1) ----
  run_cheb_layer<FEAT_IN, FEAT_HID>(x, w1t, b1, h, bufA, bufB, bufC, abf,
                                    row, col, nrm, /*relu=*/1, stream);

  // ---- layer 2: out = cheb(h, W2) + b2 ----
  run_cheb_layer<FEAT_HID, FEAT_OUT>(h, w2t, b2, out, bufA, bufB, bufC, abf,
                                     row, col, nrm, /*relu=*/0, stream);
}
